// DistanceAwareTransformerEncoderLayer_53695681134865
// MI455X (gfx1250) — compile-verified, matrix operand order checked by reading, not run on hardware
//
#include <hip/hip_runtime.h>
#include <hip/hip_bf16.h>
#include <cstdint>

// ---------------------------------------------------------------------------
// Shapes (compile-time constants for this module)
// ---------------------------------------------------------------------------
static constexpr int BB = 4;      // batch
static constexpr int LL = 2048;   // sequence length
static constexpr int EE = 512;    // embed dim
static constexpr int HH = 8;      // heads
static constexpr int HDIM = 64;   // head dim
static constexpr int FF = 2048;   // ffn dim
static constexpr int ROWS = BB * LL;  // 8192

typedef __attribute__((ext_vector_type(16))) __bf16 v16bf;
typedef __attribute__((ext_vector_type(8)))  float  v8f;
typedef __attribute__((ext_vector_type(4)))  unsigned int v4u;
typedef __attribute__((ext_vector_type(8)))  int v8i;
typedef __attribute__((ext_vector_type(4)))  int v4i;

union FragBF { v16bf v; unsigned int u[8]; };
union FragF  { v8f   v; float        f[8]; };

// K-pair base for VGPR v of a 16-bit A/B WMMA fragment (ISA 7.12.2)
__device__ __forceinline__ int kbase_of(int v, int hl) {
  return (v < 4 ? 2 * v : 16 + 2 * (v - 4)) + 8 * hl;
}

__device__ __forceinline__ unsigned pack_bf16pair(float a, float b) {
  union { __bf16 h[2]; unsigned u; } p;
  p.h[0] = (__bf16)a; p.h[1] = (__bf16)b;
  return p.u;
}

// ---------------------------------------------------------------------------
// TDM: stage one 128(rows) x 32(cols) bf16 tile from global (row stride =
// row_stride_elems bf16) into LDS with a padded row stride of 36 bf16
// (72 B = 18 dwords = 16 data dwords + 2 pad dwords).
//   pad_interval code 3 -> pad after every 16 dwords
//   pad_amount   code 1 -> insert 2 dwords of padding
// Descriptor packing per CDNA5 ISA 8.3/8.4 (D# group0 + group1, 2-D tile).
// ---------------------------------------------------------------------------
__device__ __forceinline__ void tdm_tile_128x32_bf16(void* lds_dst,
                                                     const __bf16* gsrc,
                                                     int row_stride_elems) {
  unsigned long long ga = (unsigned long long)(uintptr_t)gsrc;
  v4u g0;
  g0[0] = 1u;                                       // count=1, user mode
  g0[1] = (unsigned)(uintptr_t)lds_dst;             // lds_addr (byte offset)
  g0[2] = (unsigned)(ga & 0xFFFFFFFFull);           // global_addr[31:0]
  g0[3] = (unsigned)((ga >> 32) & 0x01FFFFFFull)    // global_addr[56:32]
        | (2u << 30);                               // type=2 ("image")
  v8i g1;
  g1[0] = (int)((1u << 16)                          // data_size=1 (2 bytes)
              | (1u << 20)                          // pad_enable
              | (3u << 22)                          // pad_interval: 16 dwords
              | (1u << 25));                        // pad_amount: 2 dwords
  g1[1] = (int)(((unsigned)row_stride_elems & 0xFFFFu) << 16); // tensor_dim0 lo16
  g1[2] = (int)(0x4000u << 16);                     // tensor_dim0 hi16=0, tensor_dim1 lo16=16384
  g1[3] = (int)(32u << 16);                         // tensor_dim1 hi16=0, tile_dim0=32
  g1[4] = (int)(128u);                              // tile_dim1=128, tile_dim2=0
  g1[5] = (int)(unsigned)row_stride_elems;          // tensor_dim0_stride[31:0]
  g1[6] = 0;                                        // stride0 hi, stride1 lo
  g1[7] = 0;                                        // stride1 hi
  v4i z4 = {0, 0, 0, 0};
#if defined(__clang_major__) && (__clang_major__ >= 23)
  v8i z8 = {0, 0, 0, 0, 0, 0, 0, 0};
  __builtin_amdgcn_tensor_load_to_lds(g0, g1, z4, z4, z8, 0);
#else
  __builtin_amdgcn_tensor_load_to_lds(g0, g1, z4, z4, 0);
#endif
}

// ---------------------------------------------------------------------------
// Converters
// ---------------------------------------------------------------------------
__global__ __launch_bounds__(256)
void k_cvt_bf16(const float* __restrict__ in, __bf16* __restrict__ out, int n) {
  int i = blockIdx.x * 256 + threadIdx.x;
  if (i < n) out[i] = (__bf16)in[i];
}

// src [B,L,E] f32  ->  srcb [B,L,E] bf16  and  srcT [B,E,L] bf16
__global__ __launch_bounds__(256)
void k_cvt_transpose(const float* __restrict__ src,
                     __bf16* __restrict__ srcb, __bf16* __restrict__ srcT) {
  int i = blockIdx.x * 256 + threadIdx.x;
  if (i >= BB * LL * EE) return;
  int e = i & (EE - 1);
  int l = (i >> 9) & (LL - 1);
  int b = i >> 20;
  float v = src[i];
  srcb[i] = (__bf16)v;
  srcT[((size_t)b * EE + e) * LL + l] = (__bf16)v;
}

// ---------------------------------------------------------------------------
// Generic bf16 GEMM:  C[M,N] = A[M,K] * B[N,K]^T + bias  (optional relu)
// block tile 128x128, 8 waves, each wave 64x32 (4x2 WMMA tiles), K step 32.
// Tiles staged by the Tensor Data Mover (wave 0 issues, TENSORcnt + barrier),
// double-buffered in LDS.
// ---------------------------------------------------------------------------
__global__ __launch_bounds__(256)
void k_gemm_bf16_nt(const __bf16* __restrict__ A, const __bf16* __restrict__ B,
                    const float* __restrict__ bias,
                    float* __restrict__ Cf, __bf16* __restrict__ Cb,
                    int M, int N, int K, int relu) {
  __shared__ __bf16 As[2][128][36];
  __shared__ __bf16 Bs[2][128][36];
  const int tid = threadIdx.x;
  const int lane = tid & 31, wave = tid >> 5;
  const int wy = wave >> 2, wx = wave & 3;      // 2 x 4 wave grid
  const int bm = blockIdx.y * 128, bn = blockIdx.x * 128;
  const int hl = lane >> 4, mr = lane & 15;

  FragF acc[4][2];
#pragma unroll
  for (int i = 0; i < 4; ++i)
#pragma unroll
    for (int j = 0; j < 2; ++j)
#pragma unroll
      for (int r = 0; r < 8; ++r) acc[i][j].f[r] = 0.0f;

  const int nk = K >> 5;               // K / 32 tiles

  if (wave == 0) {                     // prime the pipeline: tile 0 -> buffer 0
    tdm_tile_128x32_bf16(&As[0][0][0], A + (size_t)bm * K, K);
    tdm_tile_128x32_bf16(&Bs[0][0][0], B + (size_t)bn * K, K);
  }

  for (int t = 0; t < nk; ++t) {
    const int cur = t & 1;
    if (wave == 0) {
      if (t + 1 < nk) {                // issue next pair into other buffer
        tdm_tile_128x32_bf16(&As[cur ^ 1][0][0], A + (size_t)bm * K + (t + 1) * 32, K);
        tdm_tile_128x32_bf16(&Bs[cur ^ 1][0][0], B + (size_t)bn * K + (t + 1) * 32, K);
        __builtin_amdgcn_s_wait_tensorcnt(2);   // current pair has landed
      } else {
        __builtin_amdgcn_s_wait_tensorcnt(0);
      }
    }
    __syncthreads();                   // data visible to all waves

    FragBF af[4], bfr[2];
#pragma unroll
    for (int i = 0; i < 4; ++i) {
      int row = wy * 64 + i * 16 + mr;
#pragma unroll
      for (int v = 0; v < 8; ++v)
        af[i].u[v] = *(const unsigned*)&As[cur][row][kbase_of(v, hl)];
    }
#pragma unroll
    for (int j = 0; j < 2; ++j) {
      int row = wx * 32 + j * 16 + mr;
#pragma unroll
      for (int v = 0; v < 8; ++v)
        bfr[j].u[v] = *(const unsigned*)&Bs[cur][row][kbase_of(v, hl)];
    }
#pragma unroll
    for (int i = 0; i < 4; ++i)
#pragma unroll
      for (int j = 0; j < 2; ++j)
        acc[i][j].v = __builtin_amdgcn_wmma_f32_16x16x32_bf16(
            false, af[i].v, false, bfr[j].v, (short)0, acc[i][j].v, false, false);
    __syncthreads();                   // all reads done before buffer reuse
  }

  // epilogue: bias (+relu) then store f32 and/or bf16
#pragma unroll
  for (int i = 0; i < 4; ++i)
#pragma unroll
    for (int j = 0; j < 2; ++j) {
      int gn = bn + wx * 32 + j * 16 + mr;
      float bv = bias ? bias[gn] : 0.0f;
#pragma unroll
      for (int r = 0; r < 8; ++r) {
        int gm = bm + wy * 64 + i * 16 + r + 8 * hl;
        float v = acc[i][j].f[r] + bv;
        if (relu) v = v > 0.0f ? v : 0.0f;
        if (Cf) Cf[(size_t)gm * N + gn] = v;
        if (Cb) Cb[(size_t)gm * N + gn] = (__bf16)v;
      }
    }
}

// ---------------------------------------------------------------------------
// Fused attention: per (b, q-tile of 16 rows)
//  - loops heads: S = Q_h K_h^T (WMMA) -> LDS; row softmax; accumulate mean_h
//  - distance-bias reweight + renormalize (row Z)
//  - attn_out = A_norm @ src (WMMA, B-frags from transposed bf16 src)
//  - residual + LayerNorm1 -> X (f32) and Xb (bf16)
// Dynamic LDS: S[16][2048] f32 + Aacc[16][2048] f32 = 256 KB (fits 320 KB WGP)
// ---------------------------------------------------------------------------
__global__ __launch_bounds__(256)
void k_attn(const __bf16* __restrict__ QK,    // [B*L, 2E] bf16 (Q cols 0..511, K cols 512..1023)
            const __bf16* __restrict__ srcT,  // [B, E, L] bf16
            const float*  __restrict__ src,   // [B, L, E] f32
            const float*  __restrict__ dist,  // [B, L, L] f32
            const float*  __restrict__ dscale_p,
            const float*  __restrict__ g1, const float* __restrict__ be1,
            float* __restrict__ X, __bf16* __restrict__ Xb) {
  extern __shared__ float smem[];
  float* S    = smem;                 // [16][2048]
  float* Aacc = smem + 16 * LL;       // [16][2048]
  __shared__ float zinv[16];

  const int tid = threadIdx.x;
  const int lane = tid & 31, wave = tid >> 5;
  const int b  = blockIdx.x >> 7;     // L/16 = 128 q-tiles per batch
  const int qt = blockIdx.x & 127;
  const int qbase = qt * 16;
  const size_t rowQ = (size_t)b * LL + qbase;
  const int hl = lane >> 4, mr = lane & 15;

  for (int i = tid; i < 16 * LL; i += 256) Aacc[i] = 0.0f;

  // prefetch the 16 contiguous distance rows (128 KB) consumed late in the block
  {
    const float* dbase = dist + ((size_t)b * LL + qbase) * LL;
    for (int i = tid; i < (16 * LL) / 64; i += 256)
      __builtin_prefetch(dbase + (size_t)i * 64, 0, 1);
  }
  __syncthreads();

  // ---- per-head scores + softmax, accumulate head average into Aacc ----
  for (int h = 0; h < HH; ++h) {
    FragBF aq[2];
#pragma unroll
    for (int ks = 0; ks < 2; ++ks)
#pragma unroll
      for (int v = 0; v < 8; ++v) {
        int kk = h * HDIM + ks * 32 + kbase_of(v, hl);
        aq[ks].u[v] = *(const unsigned*)(QK + (rowQ + mr) * (2 * EE) + kk);
      }

    for (int t = 0; t < 16; ++t) {            // 8 waves x 16 = 128 n-tiles
      int nt = wave * 16 + t;
      FragF c;
#pragma unroll
      for (int r = 0; r < 8; ++r) c.f[r] = 0.0f;
#pragma unroll
      for (int ks = 0; ks < 2; ++ks) {
        FragBF bk;
#pragma unroll
        for (int v = 0; v < 8; ++v) {
          int kk = EE + h * HDIM + ks * 32 + kbase_of(v, hl);
          bk.u[v] = *(const unsigned*)(QK + ((size_t)b * LL + nt * 16 + mr) * (2 * EE) + kk);
        }
        c.v = __builtin_amdgcn_wmma_f32_16x16x32_bf16(
            false, aq[ks].v, false, bk.v, (short)0, c.v, false, false);
      }
#pragma unroll
      for (int r = 0; r < 8; ++r) {
        int m = r + 8 * hl;
        S[m * LL + nt * 16 + mr] = c.f[r] * 0.125f;   // 1/sqrt(64)
      }
    }
    __syncthreads();

    for (int rr = wave; rr < 16; rr += 8) {   // wave32 row softmax
      float* row = S + rr * LL;
      float mx = -3.0e38f;
      for (int i = lane; i < LL; i += 32) mx = fmaxf(mx, row[i]);
      for (int off = 16; off; off >>= 1) mx = fmaxf(mx, __shfl_xor(mx, off, 32));
      float sum = 0.0f;
      for (int i = lane; i < LL; i += 32) { float e = __expf(row[i] - mx); row[i] = e; sum += e; }
      for (int off = 16; off; off >>= 1) sum += __shfl_xor(sum, off, 32);
      float inv = 0.125f / sum;               // 1/H head average folded in
      float* arow = Aacc + rr * LL;
      for (int i = lane; i < LL; i += 32) arow[i] += row[i] * inv;
    }
    __syncthreads();
  }

  // ---- distance bias reweight + row renormalization ----
  const float ds = *dscale_p;
  for (int rr = wave; rr < 16; rr += 8) {
    const float* drow = dist + ((size_t)b * LL + qbase + rr) * LL;
    float dmx = -3.0e38f;
    for (int i = lane; i < LL; i += 32) dmx = fmaxf(dmx, drow[i]);
    for (int off = 16; off; off >>= 1) dmx = fmaxf(dmx, __shfl_xor(dmx, off, 32));
    float rinv = 1.0f / dmx;
    float* arow = Aacc + rr * LL;
    float z = 0.0f;
    for (int i = lane; i < LL; i += 32) {
      float bias = __expf(ds * (1.0f - drow[i] * rinv));
      float v = arow[i] * bias;
      arow[i] = v; z += v;
    }
    for (int off = 16; off; off >>= 1) z += __shfl_xor(z, off, 32);
    if (lane == 0) zinv[rr] = 1.0f / z;
  }
  __syncthreads();

  // ---- attn_out[16,512] = A_norm @ src  (WMMA over K=2048) ----
  FragF oacc[4];
#pragma unroll
  for (int j = 0; j < 4; ++j)
#pragma unroll
    for (int r = 0; r < 8; ++r) oacc[j].f[r] = 0.0f;

  const float zi = zinv[mr];                  // A-frag row scale (lane row = mr)
  for (int kb = 0; kb < LL; kb += 32) {
    FragBF afr;
#pragma unroll
    for (int v = 0; v < 8; ++v) {
      int kk = kb + kbase_of(v, hl);
      afr.u[v] = pack_bf16pair(Aacc[mr * LL + kk] * zi, Aacc[mr * LL + kk + 1] * zi);
    }
#pragma unroll
    for (int j = 0; j < 4; ++j) {
      int ecol = wave * 64 + j * 16 + mr;     // this lane's output column
      FragBF bfr;
#pragma unroll
      for (int v = 0; v < 8; ++v) {
        int kk = kb + kbase_of(v, hl);
        bfr.u[v] = *(const unsigned*)(srcT + ((size_t)b * EE + ecol) * LL + kk);
      }
      oacc[j].v = __builtin_amdgcn_wmma_f32_16x16x32_bf16(
          false, afr.v, false, bfr.v, (short)0, oacc[j].v, false, false);
    }
  }
  __syncthreads();

  // stash attn_out tile into S (reused as x_tile [16][512])
#pragma unroll
  for (int j = 0; j < 4; ++j) {
    int gn = wave * 64 + j * 16 + mr;
#pragma unroll
    for (int r = 0; r < 8; ++r) {
      int m = r + 8 * hl;
      S[m * EE + gn] = oacc[j].f[r];
    }
  }
  __syncthreads();

  // ---- residual + LayerNorm1 ----
  for (int rr = wave; rr < 16; rr += 8) {
    const float* srow = src + ((size_t)b * LL + qbase + rr) * EE;
    float* xrow = S + rr * EE;
    float s1 = 0.0f, s2 = 0.0f;
    for (int i = lane; i < EE; i += 32) {
      float v = xrow[i] + srow[i];
      xrow[i] = v; s1 += v; s2 += v * v;
    }
    for (int off = 16; off; off >>= 1) { s1 += __shfl_xor(s1, off, 32); s2 += __shfl_xor(s2, off, 32); }
    float mean = s1 * (1.0f / EE);
    float var  = s2 * (1.0f / EE) - mean * mean;
    float rstd = rsqrtf(var + 1e-5f);
    size_t orow = ((size_t)b * LL + qbase + rr) * EE;
    for (int i = lane; i < EE; i += 32) {
      float v = (xrow[i] - mean) * rstd * g1[i] + be1[i];
      X[orow + i] = v;
      Xb[orow + i] = (__bf16)v;
    }
  }
}

// ---------------------------------------------------------------------------
// out = LayerNorm(X + F2) * g2 + be2     (one wave per row, 8 rows / block)
// ---------------------------------------------------------------------------
__global__ __launch_bounds__(256)
void k_final_ln(const float* __restrict__ X, const float* __restrict__ F2,
                const float* __restrict__ g2, const float* __restrict__ be2,
                float* __restrict__ out) {
  const int row = blockIdx.x * 8 + (threadIdx.x >> 5);
  const int lane = threadIdx.x & 31;
  const float* xr = X  + (size_t)row * EE;
  const float* fr = F2 + (size_t)row * EE;
  float vals[16];
  float s1 = 0.0f, s2 = 0.0f;
#pragma unroll
  for (int i = 0; i < 16; ++i) {
    float v = xr[lane + i * 32] + fr[lane + i * 32];
    vals[i] = v; s1 += v; s2 += v * v;
  }
  for (int off = 16; off; off >>= 1) { s1 += __shfl_xor(s1, off, 32); s2 += __shfl_xor(s2, off, 32); }
  float mean = s1 * (1.0f / EE);
  float var  = s2 * (1.0f / EE) - mean * mean;
  float rstd = rsqrtf(var + 1e-5f);
  float* orow = out + (size_t)row * EE;
#pragma unroll
  for (int i = 0; i < 16; ++i) {
    int c = lane + i * 32;
    orow[c] = (vals[i] - mean) * rstd * g2[c] + be2[c];
  }
}

// ---------------------------------------------------------------------------
// Host-side orchestration
// ---------------------------------------------------------------------------
extern "C" void kernel_launch(void* const* d_in, const int* in_sizes, int n_in,
                              void* d_out, int out_size, void* d_ws, size_t ws_size,
                              hipStream_t stream) {
  const float* src       = (const float*)d_in[0];
  const float* distances = (const float*)d_in[1];
  const float* in_proj_w = (const float*)d_in[2];
  const float* in_proj_b = (const float*)d_in[3];
  const float* dist_scale= (const float*)d_in[4];
  const float* w1        = (const float*)d_in[5];
  const float* b1        = (const float*)d_in[6];
  const float* w2        = (const float*)d_in[7];
  const float* b2        = (const float*)d_in[8];
  const float* g1        = (const float*)d_in[9];
  const float* be1       = (const float*)d_in[10];
  const float* g2        = (const float*)d_in[11];
  const float* be2       = (const float*)d_in[12];
  float* out = (float*)d_out;

  // workspace carve-out (256B aligned)
  size_t off = 0;
  auto carve = [&](size_t bytes) {
    void* p = (char*)d_ws + off;
    off += (bytes + 255) & ~(size_t)255;
    return p;
  };
  __bf16* srcb = (__bf16*)carve((size_t)ROWS * EE * 2);       // [B*L, E]
  __bf16* srcT = (__bf16*)carve((size_t)BB * EE * LL * 2);    // [B, E, L]
  __bf16* wqk  = (__bf16*)carve((size_t)2 * EE * EE * 2);     // [1024, 512]
  __bf16* w1b  = (__bf16*)carve((size_t)FF * EE * 2);         // [2048, 512]
  __bf16* w2b  = (__bf16*)carve((size_t)EE * FF * 2);         // [512, 2048]
  __bf16* QK   = (__bf16*)carve((size_t)ROWS * 2 * EE * 2);   // [8192, 1024]
  float*  X    = (float*) carve((size_t)ROWS * EE * 4);       // LN1 output f32
  __bf16* Xb   = (__bf16*)carve((size_t)ROWS * EE * 2);       // LN1 output bf16
  __bf16* H1   = (__bf16*)carve((size_t)ROWS * FF * 2);       // ffn hidden bf16
  float*  F2   = (float*) carve((size_t)ROWS * EE * 4);       // ffn out f32
  (void)ws_size; (void)n_in; (void)in_sizes; (void)out_size;

  // 1) converters
  {
    int n = ROWS * EE;
    k_cvt_transpose<<<(n + 255) / 256, 256, 0, stream>>>(src, srcb, srcT);
  }
  { int n = 2 * EE * EE; k_cvt_bf16<<<(n + 255) / 256, 256, 0, stream>>>(in_proj_w, wqk, n); }
  { int n = FF * EE;     k_cvt_bf16<<<(n + 255) / 256, 256, 0, stream>>>(w1, w1b, n); }
  { int n = EE * FF;     k_cvt_bf16<<<(n + 255) / 256, 256, 0, stream>>>(w2, w2b, n); }

  // 2) QK projection: [8192,512] x [1024,512]^T -> QK bf16
  {
    dim3 grid((2 * EE) / 128, ROWS / 128);
    k_gemm_bf16_nt<<<grid, 256, 0, stream>>>(srcb, wqk, in_proj_b,
                                             nullptr, QK, ROWS, 2 * EE, EE, 0);
  }

  // 3) fused attention + LN1 (256 KB dynamic LDS per block)
  {
    size_t smem = (size_t)2 * 16 * LL * sizeof(float);
    k_attn<<<BB * (LL / 16), 256, smem, stream>>>(QK, srcT, src, distances, dist_scale,
                                                  g1, be1, X, Xb);
  }

  // 4) FFN1: relu(Xb @ w1^T + b1) -> H1 bf16
  {
    dim3 grid(FF / 128, ROWS / 128);
    k_gemm_bf16_nt<<<grid, 256, 0, stream>>>(Xb, w1b, b1, nullptr, H1, ROWS, FF, EE, 1);
  }

  // 5) FFN2: H1 @ w2^T + b2 -> F2 f32
  {
    dim3 grid(EE / 128, ROWS / 128);
    k_gemm_bf16_nt<<<grid, 256, 0, stream>>>(H1, w2b, b2, F2, nullptr, ROWS, EE, FF, 0);
  }

  // 6) out = LN(X + F2)
  k_final_ln<<<ROWS / 8, 256, 0, stream>>>(X, F2, g2, be2, out);
}
